// Feat_aggr_18425409699924
// MI455X (gfx1250) — matrix operand order repaired and checked by reading, not compiled, not run on hardware
//
#include <hip/hip_runtime.h>
#include <hip/hip_bf16.h>
#include <stdint.h>

// ---------------------------------------------------------------------------
// Problem dims (fixed by the reference): B=16, T=32, N=64 -> M=2048, C=256
// ---------------------------------------------------------------------------
#define BATCH 16
#define MDIM  2048
#define CDIM  256
#define BMROWS (BATCH * MDIM)   // 32768

typedef unsigned short u16;
typedef __attribute__((ext_vector_type(16))) __bf16 v16bf;
typedef __attribute__((ext_vector_type(8)))  __bf16 v8bf;
typedef __attribute__((ext_vector_type(8)))  float  v8f;

#if defined(__has_builtin)
#  define HAS_ASYNC_LDS  __has_builtin(__builtin_amdgcn_global_load_async_to_lds_b128)
#  define HAS_WAIT_ASYNC __has_builtin(__builtin_amdgcn_s_wait_asynccnt)
#else
#  define HAS_ASYNC_LDS  0
#  define HAS_WAIT_ASYNC 0
#endif

#if defined(__AMDGCN__) && !HAS_ASYNC_LDS
#warning "gfx1250 async-to-LDS builtins unavailable: falling back to synchronous staging"
#endif

static __device__ __forceinline__ u16 f2bf(float f) {
    __bf16 h = (__bf16)f;
    return __builtin_bit_cast(u16, h);
}

static __device__ __forceinline__ v16bf join16(v8bf lo, v8bf hi) {
    v16bf r;
#pragma unroll
    for (int i = 0; i < 8; ++i) { r[i] = lo[i]; r[i + 8] = hi[i]; }
    return r;
}

// A-fragment (16 rows x 32 K, bf16): lane L holds row (row0 + L%16);
// element j<8  -> K = k0 + (L/16)*8 + j        (16B chunk)
// element j>=8 -> K = k0 + 16 + (L/16)*8 + j-8 (16B chunk at +16 elems)
static __device__ __forceinline__ v16bf load_a_frag(const u16* base, int ld,
                                                    int row0, int k0, int lane) {
    const int r = row0 + (lane & 15);
    const int half = lane >> 4;
    const u16* p = base + (size_t)r * ld + k0 + half * 8;
    return join16(*(const v8bf*)p, *(const v8bf*)(p + 16));
}

// B-fragment (32 K x 16 cols, bf16) from storage where each output column's
// K-run is contiguous (row index = column, stride ld):
// lane L holds col (c0 + L%16); element j<8 -> K = k0 + (L/16)*8 + j,
// element j>=8 -> K = k0 + 16 + (L/16)*8 + j-8 (matches A-fragment)
static __device__ __forceinline__ v16bf load_b_frag(const u16* base, int ld,
                                                    int c0, int k0, int lane) {
    const int cc = c0 + (lane & 15);
    const int half = lane >> 4;
    const u16* p = base + (size_t)cc * ld + k0 + half * 8;
    return join16(*(const v8bf*)p, *(const v8bf*)(p + 16));
}

static __device__ __forceinline__ v8f wmma_bf16(v16bf a, v16bf b, v8f c) {
    return __builtin_amdgcn_wmma_f32_16x16x32_bf16(false, a, false, b,
                                                   (short)0, c, false, false);
}

// 16B async copy global -> LDS (per lane); IOFFSET applies to both addresses.
// Builtin signature (probe-confirmed by round-2 diagnostic):
//   (int4 addrspace(1)*, int4 addrspace(3)*, imm offset, imm cpol)
typedef int v4i_t __attribute__((vector_size(16)));
typedef __attribute__((address_space(1))) v4i_t gas_v4i;
typedef __attribute__((address_space(3))) v4i_t las_v4i;

template <int OFF>
static __device__ __forceinline__ void async_cp16(const u16* g, u16* l) {
#if HAS_ASYNC_LDS
    __builtin_amdgcn_global_load_async_to_lds_b128(
        (gas_v4i*)(uintptr_t)g, (las_v4i*)l, OFF, 0);
#else
    *(uint4*)((char*)l + OFF) = *(const uint4*)((const char*)g + OFF);
#endif
}

template <int N>
static __device__ __forceinline__ void wait_async() {
#if HAS_WAIT_ASYNC
    __builtin_amdgcn_s_wait_asynccnt(N);
#elif HAS_ASYNC_LDS
    asm volatile("s_wait_asynccnt %0" :: "i"(N) : "memory");
#endif
}

// ---------------------------------------------------------------------------
// Kernel 1: weight conversion to bf16 B-operand layouts.
//   WqT[d][c] = adj_w[c][d]      (Q = nf @ adj_w      -> B[c][d] = adj_w[c][d])
//   Wv [d][c] = affine_w[d][c]   (V = x @ affine_w^T  -> B[c][d] = affine_w[d][c])
// ---------------------------------------------------------------------------
__global__ __launch_bounds__(256) void prep_weights_kernel(
        const float* __restrict__ adj_w, const float* __restrict__ affine_w,
        u16* __restrict__ WqT, u16* __restrict__ Wv) {
    const int d = blockIdx.x;   // output column
    const int c = threadIdx.x;  // K index
    WqT[d * CDIM + c] = f2bf(adj_w[c * CDIM + d]);
    Wv[d * CDIM + c]  = f2bf(affine_w[d * CDIM + c]);
}

// ---------------------------------------------------------------------------
// Kernel 2: per-row L2 normalize; emit K = nf (bf16) and X = x (bf16).
// One 256-thread block per row.
// ---------------------------------------------------------------------------
__global__ __launch_bounds__(256) void prep_rows_kernel(
        const float* __restrict__ x, u16* __restrict__ Kbf, u16* __restrict__ Xbf) {
    const int row = blockIdx.x;
    const int c = threadIdx.x;
    const int lane = c & 31, wave = c >> 5;
    __shared__ float wsum[8];

    const float v = x[(size_t)row * CDIM + c];
    float ss = v * v;
#pragma unroll
    for (int off = 16; off > 0; off >>= 1) ss += __shfl_xor(ss, off, 32);
    if (lane == 0) wsum[wave] = ss;
    __syncthreads();
    float tot = 0.f;
#pragma unroll
    for (int i = 0; i < 8; ++i) tot += wsum[i];
    const float scale = 1.f / fmaxf(sqrtf(tot), 1e-12f);

    Kbf[(size_t)row * CDIM + c] = f2bf(v * scale);
    Xbf[(size_t)row * CDIM + c] = f2bf(v);
}

// ---------------------------------------------------------------------------
// Kernel 3: Q = K @ WqT_oper (bf16 row-major out), V = X @ Wv_oper + bias,
// V stored TRANSPOSED per batch: Vt[b][col][m] so flash B-fragments are
// contiguous. 8 waves/block, each wave owns one 16x64 tile of both outputs.
// ---------------------------------------------------------------------------
__global__ __launch_bounds__(256) void gemm_qv_kernel(
        const u16* __restrict__ Kbf, const u16* __restrict__ Xbf,
        const u16* __restrict__ WqT, const u16* __restrict__ Wv,
        const float* __restrict__ affb,
        u16* __restrict__ Qbf, u16* __restrict__ Vt) {
    const int lane = threadIdx.x & 31, wave = threadIdx.x >> 5;
    const int job = blockIdx.x * 8 + wave;      // 8192 wave jobs
    const int rowTile = job >> 2;               // 0..2047
    const int c0 = (job & 3) * 64;
    const int half = lane >> 4, ln16 = lane & 15;
    const int r0 = rowTile * 16;

    v8f aq[4], av[4];
#pragma unroll
    for (int t = 0; t < 4; ++t)
#pragma unroll
        for (int i = 0; i < 8; ++i) { aq[t][i] = 0.f; av[t][i] = 0.f; }

#pragma unroll
    for (int kc = 0; kc < 8; ++kc) {
        const int k0 = kc * 32;
        const v16bf a1 = load_a_frag(Kbf, CDIM, r0, k0, lane);
        const v16bf a2 = load_a_frag(Xbf, CDIM, r0, k0, lane);
#pragma unroll
        for (int t = 0; t < 4; ++t) {
            const v16bf b1 = load_b_frag(WqT, CDIM, c0 + t * 16, k0, lane);
            aq[t] = wmma_bf16(a1, b1, aq[t]);
            const v16bf b2 = load_b_frag(Wv, CDIM, c0 + t * 16, k0, lane);
            av[t] = wmma_bf16(a2, b2, av[t]);
        }
    }

#pragma unroll
    for (int t = 0; t < 4; ++t) {
        const int col = c0 + t * 16 + ln16;
        const float bias = affb[col];
#pragma unroll
        for (int v = 0; v < 8; ++v) {
            const int r = r0 + v + half * 8;          // global row in [0, BMROWS)
            Qbf[(size_t)r * CDIM + col] = f2bf(aq[t][v]);
            const int b = r >> 11, m = r & (MDIM - 1);
            Vt[(size_t)b * CDIM * MDIM + (size_t)col * MDIM + m] =
                f2bf(av[t][v] + bias);
        }
    }
}

// ---------------------------------------------------------------------------
// Kernel 4: flash attention + rowsum-normalize + LayerNorm + LeakyReLU.
// Block = 256 threads = 8 waves = 4 row-groups x 2 waves; 64 rows per block.
// Each wave: S-tile 16x16 (its 16-key chunk of BK=32), P via LDS,
// P@V into its 128 O-columns. Q (16x256 bf16) pinned in registers.
// Async staging: issue K then V async-to-LDS; wait K (asynccnt<=4), compute
// S/exp/P while V is still in flight, then wait 0 before P@V.
// ---------------------------------------------------------------------------
#define KSTR 280   // 256 + 24 pad (keeps 16B align, spreads banks)
#define VSTR 40    // 32 + 8 pad
#define PSTR 40

__global__ __launch_bounds__(256) void flash_kernel(
        const u16* __restrict__ Kbf, const u16* __restrict__ Qbf,
        const u16* __restrict__ Vt,
        const float* __restrict__ ln_g, const float* __restrict__ ln_b,
        float* __restrict__ out) {
    __shared__ __align__(16) u16 Klds[32 * KSTR];
    __shared__ __align__(16) u16 Vlds[256 * VSTR];
    __shared__ __align__(16) u16 Plds[4][16 * PSTR];
    __shared__ float Red[4][2][16][2];

    const int tid = threadIdx.x;
    const int lane = tid & 31, wave = tid >> 5;
    const int g = wave >> 1, w2 = wave & 1;
    const int half = lane >> 4, ln16 = lane & 15;
    const int bb = blockIdx.y;
    const int rowBase = blockIdx.x * 64;
    const int r0g = bb * MDIM + rowBase + g * 16;   // Q/K global row base

    // Q fragments, resident for the whole key loop (64 VGPRs)
    v16bf qf[8];
#pragma unroll
    for (int kc = 0; kc < 8; ++kc)
        qf[kc] = load_a_frag(Qbf, CDIM, r0g, kc * 32, lane);

    v8f acc[8];
#pragma unroll
    for (int t = 0; t < 8; ++t)
#pragma unroll
        for (int i = 0; i < 8; ++i) acc[t][i] = 0.f;
    float sumP[8];
#pragma unroll
    for (int v = 0; v < 8; ++v) sumP[v] = 0.f;

    // per-thread staging coordinates
    const int krow = tid >> 3, cseg = (tid & 7) * 32;   // K: 32 rows x 256
    const int vcol = tid;                               // V: 256 cols x 32 keys

    for (int kb = 0; kb < MDIM / 32; ++kb) {
        const int kbase = kb * 32;
        __syncthreads();   // previous iteration's LDS reads complete
        {
            const u16* ks = Kbf + (size_t)(bb * MDIM + kbase + krow) * CDIM + cseg;
            u16* kd = &Klds[krow * KSTR + cseg];
            async_cp16<0>(ks, kd);  async_cp16<16>(ks, kd);
            async_cp16<32>(ks, kd); async_cp16<48>(ks, kd);
            const u16* vs = Vt + (size_t)bb * CDIM * MDIM + (size_t)vcol * MDIM + kbase;
            u16* vd = &Vlds[vcol * VSTR];
            async_cp16<0>(vs, vd);  async_cp16<16>(vs, vd);
            async_cp16<32>(vs, vd); async_cp16<48>(vs, vd);
        }
        wait_async<4>();   // K chunks complete (async loads retire in order)
        __syncthreads();   // all waves' K staged (V may still be in flight)

        // S = Q K^T for this wave's 16-key chunk (keys w2*16 .. w2*16+15)
        v8f s;
#pragma unroll
        for (int i = 0; i < 8; ++i) s[i] = 0.f;
#pragma unroll
        for (int kc = 0; kc < 8; ++kc) {
            const v16bf bk = load_b_frag(Klds, KSTR, w2 * 16, kc * 32, lane);
            s = wmma_bf16(qf[kc], bk, s);
        }

        // P = exp(5*S)  (softmax(5(A-max)) == softmax(5A); 5S is O(10), safe)
#pragma unroll
        for (int v = 0; v < 8; ++v) {
            const float e = __expf(5.f * s[v]);
            sumP[v] += e;
            Plds[g][(v + half * 8) * PSTR + w2 * 16 + ln16] = f2bf(e);
        }
        wait_async<0>();   // V staged
        __syncthreads();   // both waves' P chunks + V visible

        // O += P (16x32) @ V (32 x 128-col slice)
        const v16bf pa = load_a_frag(&Plds[g][0], PSTR, 0, 0, lane);
#pragma unroll
        for (int t = 0; t < 8; ++t) {
            const v16bf bv = load_b_frag(Vlds, VSTR, w2 * 128 + t * 16, 0, lane);
            acc[t] = wmma_bf16(pa, bv, acc[t]);
        }
    }

    // ---- rowsum across the 16 lanes of each half, then across the 2 waves
#pragma unroll
    for (int v = 0; v < 8; ++v)
#pragma unroll
        for (int off = 1; off < 16; off <<= 1)
            sumP[v] += __shfl_xor(sumP[v], off, 16);
    __syncthreads();
    if (ln16 == 0)
#pragma unroll
        for (int v = 0; v < 8; ++v) Red[g][w2][v + half * 8][0] = sumP[v];
    __syncthreads();
    float inv[8];
#pragma unroll
    for (int v = 0; v < 8; ++v) {
        const int r = v + half * 8;
        inv[v] = 1.f / (Red[g][0][r][0] + Red[g][1][r][0]);
    }
    __syncthreads();   // Red slots reused for LN stats

    // ---- normalize by rowsum, accumulate LN partials
    float sm[8], sq[8];
#pragma unroll
    for (int v = 0; v < 8; ++v) { sm[v] = 0.f; sq[v] = 0.f; }
#pragma unroll
    for (int t = 0; t < 8; ++t)
#pragma unroll
        for (int v = 0; v < 8; ++v) {
            const float o = acc[t][v] * inv[v];
            acc[t][v] = o;
            sm[v] += o;
            sq[v] += o * o;
        }
#pragma unroll
    for (int v = 0; v < 8; ++v)
#pragma unroll
        for (int off = 1; off < 16; off <<= 1) {
            sm[v] += __shfl_xor(sm[v], off, 16);
            sq[v] += __shfl_xor(sq[v], off, 16);
        }
    if (ln16 == 0)
#pragma unroll
        for (int v = 0; v < 8; ++v) {
            Red[g][w2][v + half * 8][0] = sm[v];
            Red[g][w2][v + half * 8][1] = sq[v];
        }
    __syncthreads();

    float gcol[8], bcol[8];
#pragma unroll
    for (int t = 0; t < 8; ++t) {
        const int col = w2 * 128 + t * 16 + ln16;
        gcol[t] = ln_g[col];
        bcol[t] = ln_b[col];
    }

#pragma unroll
    for (int v = 0; v < 8; ++v) {
        const int r = v + half * 8;
        const float mu = (Red[g][0][r][0] + Red[g][1][r][0]) * (1.f / CDIM);
        const float ex2 = (Red[g][0][r][1] + Red[g][1][r][1]) * (1.f / CDIM);
        const float rstd = rsqrtf(fmaxf(ex2 - mu * mu, 0.f) + 1e-5f);
        const int rowOut = rowBase + g * 16 + r;
#pragma unroll
        for (int t = 0; t < 8; ++t) {
            const int col = w2 * 128 + t * 16 + ln16;
            float h = (acc[t][v] - mu) * rstd * gcol[t] + bcol[t];
            h = (h >= 0.f) ? h : 0.01f * h;
            out[((size_t)bb * MDIM + rowOut) * CDIM + col] = h;
        }
    }
}

// ---------------------------------------------------------------------------
extern "C" void kernel_launch(void* const* d_in, const int* in_sizes, int n_in,
                              void* d_out, int out_size, void* d_ws, size_t ws_size,
                              hipStream_t stream) {
    (void)in_sizes; (void)n_in; (void)out_size; (void)ws_size;
    const float* local_feat = (const float*)d_in[0];
    // d_in[1] (global_feat) and d_in[2] (pos) are unused by the reference.
    const float* adj_w    = (const float*)d_in[3];
    const float* affine_w = (const float*)d_in[4];
    const float* affine_b = (const float*)d_in[5];
    const float* ln_g     = (const float*)d_in[6];
    const float* ln_b     = (const float*)d_in[7];
    float* out = (float*)d_out;

    // workspace carve-up (bf16 halves): K, X, Q, Vt (each BM*C), WqT, Wv
    u16* ws = (u16*)d_ws;
    const size_t NE = (size_t)BMROWS * CDIM;      // 8,388,608 elems
    u16* Kbf = ws;
    u16* Xbf = ws + NE;
    u16* Qbf = ws + 2 * NE;
    u16* Vt  = ws + 3 * NE;
    u16* WqT = ws + 4 * NE;
    u16* Wv  = WqT + CDIM * CDIM;

    prep_weights_kernel<<<dim3(CDIM), dim3(CDIM), 0, stream>>>(adj_w, affine_w, WqT, Wv);
    prep_rows_kernel<<<dim3(BMROWS), dim3(256), 0, stream>>>(local_feat, Kbf, Xbf);
    gemm_qv_kernel<<<dim3(BMROWS / 16 * 4 / 8), dim3(256), 0, stream>>>(
        Kbf, Xbf, WqT, Wv, affine_b, Qbf, Vt);
    flash_kernel<<<dim3(MDIM / 64, BATCH), dim3(256), 0, stream>>>(
        Kbf, Qbf, Vt, ln_g, ln_b, out);
}
